// LearnableGateCAPM_33715493273650
// MI455X (gfx1250) — matrix operand order repaired and checked by reading, not compiled, not run on hardware
//
#include <hip/hip_runtime.h>
#include <math.h>

// ---------------------------------------------------------------------------
// Problem constants (match reference)
// ---------------------------------------------------------------------------
#define NPAIR 512
#define NCLS  51
#define HDIM  512
#define EDIM  768
#define PDIM  4096
#define NOBJ  151
#define TWO_H 1024
#define THREE_E 2304

typedef __bf16 bf16;
typedef __bf16  v16bf __attribute__((ext_vector_type(16)));
typedef float   v8f   __attribute__((ext_vector_type(8)));
typedef int     v4i   __attribute__((vector_size(16)));   // async b128 payload type

// LDS tile stride (bf16 elems): 32 K-values + 8 pad -> 80B row stride,
// keeps 16B alignment for b128 fragment reads and async 16B chunk writes.
#define LDS_S 40

// ---------------------------------------------------------------------------
// CDNA5 async global->LDS staging (ASYNCcnt), guarded by __has_builtin so the
// file still compiles (sync fallback) on toolchains without the builtins.
// ---------------------------------------------------------------------------
#if defined(__gfx1250__) && \
    __has_builtin(__builtin_amdgcn_global_load_async_to_lds_b128) && \
    __has_builtin(__builtin_amdgcn_s_wait_asynccnt)
#define USE_ASYNC 1
#define WAIT_ASYNC(n) __builtin_amdgcn_s_wait_asynccnt(n)
#else
#define USE_ASYNC 0
#define WAIT_ASYNC(n)
#endif

// Stage a 128-row x 32-col bf16 tile from row-major G (leading dim ld) at
// (r0, k0) into LDS tile L (stride LDS_S). Async path: 512 x 16B chunks,
// 2 async instructions per thread (= 2 ASYNCcnt increments per wave).
__device__ __forceinline__ void stage_tile(const bf16* __restrict__ G, size_t ld,
                                           int r0, int k0, bf16* L, int tid) {
#if USE_ASYNC
#pragma unroll
  for (int r = 0; r < 2; ++r) {
    int j   = tid + 256 * r;
    int row = j >> 2, blk = j & 3;
    __builtin_amdgcn_global_load_async_to_lds_b128(
        (__attribute__((address_space(1))) v4i*)(G + (size_t)(r0 + row) * ld + k0 + blk * 8),
        (__attribute__((address_space(3))) v4i*)(L + row * LDS_S + blk * 8),
        0, 0);
  }
#else
  for (int i = tid; i < 128 * 32; i += 256) {
    int row = i >> 5, kk = i & 31;
    L[row * LDS_S + kk] = G[(size_t)(r0 + row) * ld + k0 + kk];
  }
#endif
}

// ---------------------------------------------------------------------------
// WMMA fragment loaders (wave32 16x16x32 bf16; this file's contiguous K order, same for A and B)
// A (16x32, MxK):  lanes 0-15 -> M=lane, elems 0..15 = K 0..15
//                  lanes16-31 -> M=lane-16, elems 0..15 = K 16..31
// B (32x16, KxN):  lanes 0-15 -> N=lane, K 0..15 ; lanes16-31 -> N=lane-16, K 16..31
// C/D (16x16 f32): lane&15 = N, vgpr r -> M = r + 8*(lane>=16)
// Both operand tiles are stored K-contiguous in LDS (B is pre-transposed), so
// fragments assemble from 16B-aligned contiguous LDS reads (ds_load_b128).
// ---------------------------------------------------------------------------
__device__ __forceinline__ v16bf ld_a_frag(const bf16* As, int row0, int lane) {
  const bf16* p = As + (row0 + (lane & 15)) * LDS_S + ((lane >> 4) << 4);
  v16bf a;
#pragma unroll
  for (int e = 0; e < 8; ++e) { a[e] = p[e]; a[e + 8] = p[e + 8]; }
  return a;
}

__device__ __forceinline__ v16bf ld_b_frag(const bf16* Bt, int col0, int lane) {
  const bf16* p = Bt + (col0 + (lane & 15)) * LDS_S + ((lane >> 4) << 4);
  v16bf b;
#pragma unroll
  for (int e = 0; e < 16; ++e) b[e] = p[e];
  return b;
}

// One K=32 step for a wave owning a 32(M) x 64(N) sub-tile.
__device__ __forceinline__ void mma_step(const bf16* As, const bf16* Bt,
                                         int wm, int wn, int lane, v8f acc[2][4]) {
  v16bf a0 = ld_a_frag(As, wm * 32 + 0,  lane);
  v16bf a1 = ld_a_frag(As, wm * 32 + 16, lane);
#pragma unroll
  for (int fn = 0; fn < 4; ++fn) {
    v16bf b = ld_b_frag(Bt, wn * 64 + fn * 16, lane);
    acc[0][fn] = __builtin_amdgcn_wmma_f32_16x16x32_bf16(
        false, a0, false, b, (short)0, acc[0][fn], false, false);
    acc[1][fn] = __builtin_amdgcn_wmma_f32_16x16x32_bf16(
        false, a1, false, b, (short)0, acc[1][fn], false, false);
  }
}

__device__ __forceinline__ void zero_acc(v8f acc[2][4]) {
#pragma unroll
  for (int i = 0; i < 2; ++i)
#pragma unroll
    for (int j = 0; j < 4; ++j)
#pragma unroll
      for (int e = 0; e < 8; ++e) acc[i][j][e] = 0.0f;
}

// ---------------------------------------------------------------------------
// One-time weight prep
// ---------------------------------------------------------------------------
__global__ void k_cvt_bf16(const float* __restrict__ in, bf16* __restrict__ out, int n) {
  for (int i = blockIdx.x * blockDim.x + threadIdx.x; i < n; i += gridDim.x * blockDim.x)
    out[i] = (bf16)in[i];
}

// in: (K x N) f32 row-major  ->  out: (N x K) bf16 row-major.  K,N mult of 32.
// block (32,8), grid (N/32, K/32)
__global__ void k_transpose_cvt(const float* __restrict__ in, bf16* __restrict__ out,
                                int K, int N) {
  __shared__ float tile[32][33];
  const int tx = threadIdx.x, ty = threadIdx.y;
  const int n0 = blockIdx.x * 32, k0 = blockIdx.y * 32;
  for (int r = ty; r < 32; r += 8)
    tile[r][tx] = in[(size_t)(k0 + r) * N + (n0 + tx)];
  __syncthreads();
  for (int r = ty; r < 32; r += 8)
    out[(size_t)(n0 + r) * K + (k0 + tx)] = (bf16)tile[tx][r];
}

// ---------------------------------------------------------------------------
// GEMM1: x(N,3E) @ w1(3E,2H) + b1 -> h (f32). A generated on the fly from
// clip_obj_embeds gather (concat [e_sub, e_obj, e_sub-e_obj]); B = w1^T async.
// grid: (2H/128, N/128)
// ---------------------------------------------------------------------------
__device__ __forceinline__ void gemm1_stage_a(const float* __restrict__ clip,
                                              const int* __restrict__ pair,
                                              int m0, int k0, bf16* As, int tid) {
  for (int i = tid; i < 128 * 32; i += 256) {
    int mm = i >> 5, kk = i & 31;
    int n = m0 + mm, k = k0 + kk;
    int si = pair[2 * n];     si = si < 0 ? 0 : (si > NOBJ - 1 ? NOBJ - 1 : si);
    int oi = pair[2 * n + 1]; oi = oi < 0 ? 0 : (oi > NOBJ - 1 ? NOBJ - 1 : oi);
    float v;
    if (k < EDIM)          v = clip[si * EDIM + k];
    else if (k < 2 * EDIM) v = clip[oi * EDIM + (k - EDIM)];
    else                   v = clip[si * EDIM + (k - 2 * EDIM)] -
                               clip[oi * EDIM + (k - 2 * EDIM)];
    As[mm * LDS_S + kk] = (bf16)v;
  }
}

__global__ void k_gemm1(const float* __restrict__ clip, const int* __restrict__ pair,
                        const bf16* __restrict__ w1t, const float* __restrict__ b1,
                        float* __restrict__ h) {
  __shared__ bf16 As[2][128 * LDS_S];
  __shared__ bf16 Bt[2][128 * LDS_S];
  const int tid = threadIdx.x, lane = tid & 31, wid = tid >> 5;
  const int wm = wid >> 1, wn = wid & 1;
  const int m0 = blockIdx.y * 128, p0 = blockIdx.x * 128;
  const int nT = THREE_E / 32;

  v8f acc[2][4];
  zero_acc(acc);

  gemm1_stage_a(clip, pair, m0, 0, As[0], tid);
  stage_tile(w1t, THREE_E, p0, 0, Bt[0], tid);

  for (int t = 0; t < nT; ++t) {
    const int cur = t & 1;
    if (t + 1 < nT) {
      gemm1_stage_a(clip, pair, m0, (t + 1) * 32, As[cur ^ 1], tid);
      stage_tile(w1t, THREE_E, p0, (t + 1) * 32, Bt[cur ^ 1], tid);
      WAIT_ASYNC(2);   // tile t's 2 async issues complete (in-order)
    } else {
      WAIT_ASYNC(0);
    }
    __syncthreads();
    mma_step(As[cur], Bt[cur], wm, wn, lane, acc);
    __syncthreads();
  }

#pragma unroll
  for (int fm = 0; fm < 2; ++fm)
#pragma unroll
    for (int fn = 0; fn < 4; ++fn)
#pragma unroll
      for (int r = 0; r < 8; ++r) {
        int m = m0 + wm * 32 + fm * 16 + ((lane >> 4) << 3) + r;
        int p = p0 + wn * 64 + fn * 16 + (lane & 15);
        h[(size_t)m * TWO_H + p] = acc[fm][fn][r] + b1[p];
      }
}

// ---------------------------------------------------------------------------
// LayerNorm(2H) + ReLU, emit bf16. One block per row.
// ---------------------------------------------------------------------------
__global__ void k_ln_relu(const float* __restrict__ h, const float* __restrict__ g,
                          const float* __restrict__ b, bf16* __restrict__ out) {
  __shared__ float s1[256], s2[256];
  const int row = blockIdx.x, tid = threadIdx.x;
  float s = 0.f, ss = 0.f;
  for (int j = tid; j < TWO_H; j += 256) {
    float x = h[(size_t)row * TWO_H + j];
    s += x; ss += x * x;
  }
  s1[tid] = s; s2[tid] = ss;
  __syncthreads();
  for (int st = 128; st > 0; st >>= 1) {
    if (tid < st) { s1[tid] += s1[tid + st]; s2[tid] += s2[tid + st]; }
    __syncthreads();
  }
  float mean = s1[0] * (1.0f / TWO_H);
  float var  = s2[0] * (1.0f / TWO_H) - mean * mean;
  float rstd = rsqrtf(var + 1e-5f);
  for (int j = tid; j < TWO_H; j += 256) {
    float x = h[(size_t)row * TWO_H + j];
    float y = (x - mean) * rstd * g[j] + b[j];
    out[(size_t)row * TWO_H + j] = (bf16)fmaxf(y, 0.0f);
  }
}

// ---------------------------------------------------------------------------
// Generic double-buffered GEMM: A (bf16, MxK row-major) @ B^T-stored weights
// (bf16, NoutxK row-major) with per-kernel epilogues. Both operands async.
// ---------------------------------------------------------------------------
#define GEMM_BODY(KDIM, A_PTR, A_LD, BT_PTR, EPILOGUE)                        \
  __shared__ bf16 As[2][128 * LDS_S];                                         \
  __shared__ bf16 Bt[2][128 * LDS_S];                                         \
  const int tid = threadIdx.x, lane = tid & 31, wid = tid >> 5;               \
  const int wm = wid >> 1, wn = wid & 1;                                      \
  const int m0 = blockIdx.y * 128, p0 = blockIdx.x * 128;                     \
  const int nT = (KDIM) / 32;                                                 \
  v8f acc[2][4];                                                              \
  zero_acc(acc);                                                              \
  stage_tile((A_PTR), (A_LD), m0, 0, As[0], tid);                             \
  stage_tile((BT_PTR), (KDIM), p0, 0, Bt[0], tid);                            \
  for (int t = 0; t < nT; ++t) {                                              \
    const int cur = t & 1;                                                    \
    if (t + 1 < nT) {                                                         \
      stage_tile((A_PTR), (A_LD), m0, (t + 1) * 32, As[cur ^ 1], tid);        \
      stage_tile((BT_PTR), (KDIM), p0, (t + 1) * 32, Bt[cur ^ 1], tid);       \
      WAIT_ASYNC(4);                                                          \
    } else {                                                                  \
      WAIT_ASYNC(0);                                                          \
    }                                                                         \
    __syncthreads();                                                          \
    mma_step(As[cur], Bt[cur], wm, wn, lane, acc);                            \
    __syncthreads();                                                          \
  }                                                                           \
  _Pragma("unroll")                                                           \
  for (int fm = 0; fm < 2; ++fm)                                              \
    _Pragma("unroll")                                                         \
    for (int fn = 0; fn < 4; ++fn)                                            \
      _Pragma("unroll")                                                       \
      for (int r = 0; r < 8; ++r) {                                           \
        int m = m0 + wm * 32 + fm * 16 + ((lane >> 4) << 3) + r;              \
        int p = p0 + wn * 64 + fn * 16 + (lane & 15);                         \
        EPILOGUE;                                                             \
      }

// GEMM2: h_ln(N,2H) @ w2 + b2 -> context (f32 + bf16). grid (H/128, N/128)
__global__ void k_gemm2(const bf16* __restrict__ A, const bf16* __restrict__ w2t,
                        const float* __restrict__ b2,
                        float* __restrict__ ctxf, bf16* __restrict__ ctxb) {
  GEMM_BODY(TWO_H, A, TWO_H, w2t, {
    float v = acc[fm][fn][r] + b2[p];
    ctxf[(size_t)m * HDIM + p] = v;
    ctxb[(size_t)m * HDIM + p] = (bf16)v;
  })
}

// GEMM3: context(N,H) @ c2b_w + c2b_b -> context_proj (bf16). grid (4,4)
__global__ void k_gemm3(const bf16* __restrict__ A, const bf16* __restrict__ wt,
                        const float* __restrict__ bias, bf16* __restrict__ cp) {
  GEMM_BODY(HDIM, A, HDIM, wt, {
    cp[(size_t)m * HDIM + p] = (bf16)(acc[fm][fn][r] + bias[p]);
  })
}

// ---------------------------------------------------------------------------
// Gates: sigmoid( sum_h ctx[n,h]*pred[c,h]*gm_w[h/HD] / (sqrt(HD)*max(T,0.01)) )
// One wave per (n,c).
// ---------------------------------------------------------------------------
__global__ void k_gates(const float* __restrict__ ctx, const float* __restrict__ pred,
                        const float* __restrict__ gmw, const float* __restrict__ temp,
                        float* __restrict__ gates) {
  int gid = blockIdx.x * 8 + (threadIdx.x >> 5);
  if (gid >= NPAIR * NCLS) return;
  int n = gid / NCLS, c = gid % NCLS;
  int lane = threadIdx.x & 31;
  float s = 0.f;
  for (int j = lane; j < HDIM; j += 32)
    s += ctx[(size_t)n * HDIM + j] * pred[(size_t)c * HDIM + j] * gmw[j >> 7];
#pragma unroll
  for (int off = 16; off > 0; off >>= 1) s += __shfl_xor(s, off, 32);
  if (lane == 0) {
    float t = fmaxf(temp[0], 0.01f);
    float logit = s * (1.0f / (sqrtf(128.0f) * t));
    gates[gid] = 1.0f / (1.0f + expf(-logit));
  }
}

// ---------------------------------------------------------------------------
// GEMM4 (dominant): per class c, A = context_proj ⊙ class_basis[c] (N x H)
// built in-LDS; B = dp_w^T async. Fused epilogue:
//   out[n,c,p] = proto[c,p] + mod_scale * gates[n,c] * tanh(acc + dp_b[p])
// grid: (P/128, N/128, C)
// ---------------------------------------------------------------------------
__device__ __forceinline__ void gemm4_stage_a(const bf16* __restrict__ cp,
                                              const bf16* __restrict__ cb,
                                              int c, int m0, int k0, bf16* As, int tid) {
  for (int i = tid; i < 128 * 32; i += 256) {
    int mm = i >> 5, kk = i & 31;
    float a = (float)cp[(size_t)(m0 + mm) * HDIM + (k0 + kk)];
    float w = (float)cb[(size_t)c * HDIM + (k0 + kk)];
    As[mm * LDS_S + kk] = (bf16)(a * w);
  }
}

__global__ void k_gemm4(const bf16* __restrict__ cp, const bf16* __restrict__ cb,
                        const bf16* __restrict__ dpwt, const float* __restrict__ dpb,
                        const float* __restrict__ gates, const float* __restrict__ proto,
                        const float* __restrict__ mod_scale, float* __restrict__ out) {
  __shared__ bf16 As[2][128 * LDS_S];
  __shared__ bf16 Bt[2][128 * LDS_S];
  const int tid = threadIdx.x, lane = tid & 31, wid = tid >> 5;
  const int wm = wid >> 1, wn = wid & 1;
  const int c  = blockIdx.z;
  const int m0 = blockIdx.y * 128, p0 = blockIdx.x * 128;
  const int nT = HDIM / 32;

  v8f acc[2][4];
  zero_acc(acc);

  gemm4_stage_a(cp, cb, c, m0, 0, As[0], tid);
  stage_tile(dpwt, HDIM, p0, 0, Bt[0], tid);

  for (int t = 0; t < nT; ++t) {
    const int cur = t & 1;
    if (t + 1 < nT) {
      gemm4_stage_a(cp, cb, c, m0, (t + 1) * 32, As[cur ^ 1], tid);
      stage_tile(dpwt, HDIM, p0, (t + 1) * 32, Bt[cur ^ 1], tid);
      WAIT_ASYNC(2);
    } else {
      WAIT_ASYNC(0);
    }
    __syncthreads();
    mma_step(As[cur], Bt[cur], wm, wn, lane, acc);
    __syncthreads();
  }

  const float ms = mod_scale[0];
#pragma unroll
  for (int fm = 0; fm < 2; ++fm)
#pragma unroll
    for (int fn = 0; fn < 4; ++fn)
#pragma unroll
      for (int r = 0; r < 8; ++r) {
        int m = m0 + wm * 32 + fm * 16 + ((lane >> 4) << 3) + r;  // pair index n
        int p = p0 + wn * 64 + fn * 16 + (lane & 15);             // proto dim
        float d = tanhf(acc[fm][fn][r] + dpb[p]);
        float g = gates[(size_t)m * NCLS + c];
        out[((size_t)m * NCLS + c) * PDIM + p] =
            proto[(size_t)c * PDIM + p] + ms * g * d;
      }
}

// ---------------------------------------------------------------------------
// Host-side launcher
// ---------------------------------------------------------------------------
extern "C" void kernel_launch(void* const* d_in, const int* in_sizes, int n_in,
                              void* d_out, int out_size, void* d_ws, size_t ws_size,
                              hipStream_t stream) {
  (void)in_sizes; (void)n_in; (void)out_size; (void)ws_size;

  const float* proto  = (const float*)d_in[0];
  const int*   pair   = (const int*)d_in[1];
  const float* clip   = (const float*)d_in[2];
  const float* w1     = (const float*)d_in[3];
  const float* b1     = (const float*)d_in[4];
  const float* ln_g   = (const float*)d_in[5];
  const float* ln_b   = (const float*)d_in[6];
  const float* w2     = (const float*)d_in[7];
  const float* b2     = (const float*)d_in[8];
  const float* pred   = (const float*)d_in[9];
  const float* gmw    = (const float*)d_in[10];
  const float* c2bw   = (const float*)d_in[11];
  const float* c2bb   = (const float*)d_in[12];
  const float* cbasis = (const float*)d_in[13];
  const float* dpw    = (const float*)d_in[14];
  const float* dpb    = (const float*)d_in[15];
  const float* temp   = (const float*)d_in[16];
  const float* msc    = (const float*)d_in[17];
  float* out = (float*)d_out;

  // workspace carve-up (256B aligned)
  char* w = (char*)d_ws;
  auto carve = [&](size_t bytes) -> void* {
    void* p = (void*)w;
    w += (bytes + 255) & ~(size_t)255;
    return p;
  };
  bf16*  w1t    = (bf16*)carve((size_t)THREE_E * TWO_H * 2);  // (2H x 3E)
  bf16*  w2t    = (bf16*)carve((size_t)TWO_H * HDIM * 2);     // (H x 2H)
  bf16*  c2bt   = (bf16*)carve((size_t)HDIM * HDIM * 2);      // (H x H)
  bf16*  dpwt   = (bf16*)carve((size_t)HDIM * PDIM * 2);      // (P x H)
  bf16*  cbb    = (bf16*)carve((size_t)NCLS * HDIM * 2);
  float* h_f32  = (float*)carve((size_t)NPAIR * TWO_H * 4);
  bf16*  hln    = (bf16*)carve((size_t)NPAIR * TWO_H * 2);
  float* ctxf   = (float*)carve((size_t)NPAIR * HDIM * 4);
  bf16*  ctxb   = (bf16*)carve((size_t)NPAIR * HDIM * 2);
  bf16*  cpb    = (bf16*)carve((size_t)NPAIR * HDIM * 2);
  float* gates  = (float*)carve((size_t)NPAIR * NCLS * 4);

  // 1) one-time weight prep: transpose+convert so async tiles are K-contiguous
  dim3 tb(32, 8);
  k_transpose_cvt<<<dim3(TWO_H / 32, THREE_E / 32), tb, 0, stream>>>(w1,   w1t,  THREE_E, TWO_H);
  k_transpose_cvt<<<dim3(HDIM / 32,  TWO_H / 32),  tb, 0, stream>>>(w2,   w2t,  TWO_H,   HDIM);
  k_transpose_cvt<<<dim3(HDIM / 32,  HDIM / 32),   tb, 0, stream>>>(c2bw, c2bt, HDIM,    HDIM);
  k_transpose_cvt<<<dim3(PDIM / 32,  HDIM / 32),   tb, 0, stream>>>(dpw,  dpwt, HDIM,    PDIM);
  k_cvt_bf16<<<(NCLS * HDIM + 1023) / 1024, 256, 0, stream>>>(cbasis, cbb, NCLS * HDIM);

  // 2) front end
  k_gemm1<<<dim3(TWO_H / 128, NPAIR / 128), 256, 0, stream>>>(clip, pair, w1t, b1, h_f32);
  k_ln_relu<<<NPAIR, 256, 0, stream>>>(h_f32, ln_g, ln_b, hln);
  k_gemm2<<<dim3(HDIM / 128, NPAIR / 128), 256, 0, stream>>>(hln, w2t, b2, ctxf, ctxb);
  k_gates<<<(NPAIR * NCLS + 7) / 8, 256, 0, stream>>>(ctxf, pred, gmw, temp, gates);
  k_gemm3<<<dim3(HDIM / 128, NPAIR / 128), 256, 0, stream>>>(ctxb, c2bt, c2bb, cpb);

  // 3) dominant modulated projection, fully fused epilogue
  k_gemm4<<<dim3(PDIM / 128, NPAIR / 128, NCLS), 256, 0, stream>>>(
      cpb, cbb, dpwt, dpb, gates, proto, msc, out);
}
